// TrajectoryDecoder_76957224009997
// MI455X (gfx1250) — matrix operand order repaired and checked by reading, not compiled, not run on hardware
//
#include <hip/hip_runtime.h>
#include <math.h>

typedef __attribute__((ext_vector_type(16))) _Float16 v16h;
typedef __attribute__((ext_vector_type(8)))  float    v8f;

// ---------------- problem dims ----------------
#define T_STEPS      300
#define OUT_UNC_BASE 2764800   // 256*300*4*9 trajectory floats come first

// ---------------- workspace layout (uint32 indices into d_ws) ----------------
// B-fragment layout per weight matrix: [wave(8)][gate(3:i,c,o)][ktile][lane(32)][dword(8)]
#define OFF_W0   0        // KT=1 : 8*3*1*32*8 = 6144 dwords (state features only)
#define OFF_W1   6144     // KT=4 : 24576 dwords
#define OFF_W2   30720    // KT=4 : 24576 dwords
#define OFF_WH   55296    // head [ktile(4)][lane(32)][dword(8)] = 1024 dwords
#define FRAG_DWORDS 56320 // total fragment dwords (W0|W1|W2|WH contiguous) = 220 KB
#define OFF_BL   56320    // 3*512 floats (LSTM biases)
#define OFF_BH   57856    // 16 floats (fused head bias)
#define OFF_PE   57872    // 256*64 floats (physics embedding)
#define OFF_GAT  74256    // _Float16 region: [layer(3)][g/a/tr(3)][row(1024)][128]
#define OFF_TS   664080   // 300*512 floats: per-step temb contribution to layer-0 zg

// ---------------- branch-free fast transcendentals (v_exp2/v_log2/v_rcp) ----
#define LOG2E 1.44269504088896340736f
#define LN2   0.69314718055994530942f
static __device__ __forceinline__ float fexp(float x) {
  return __builtin_amdgcn_exp2f(x * LOG2E);
}
static __device__ __forceinline__ float fsigmoid(float x) {
  return __builtin_amdgcn_rcpf(1.f + fexp(-x));
}
static __device__ __forceinline__ float ftanh(float x) {
  return __builtin_fmaf(2.f, fsigmoid(2.f * x), -1.f);
}
static __device__ __forceinline__ float fsoftplus(float x) {
  float l = LN2 * __builtin_amdgcn_logf(1.f + fexp(x));
  return (x > 20.f) ? x : l;      // both sides computed -> v_cndmask, no branch
}

static __device__ __forceinline__ unsigned int pack2h(float lo, float hi) {
  union { _Float16 h[2]; unsigned int u; } p;
  p.h[0] = (_Float16)lo; p.h[1] = (_Float16)hi;
  return p.u;
}

// ============================================================================
// P1: convert weights -> f16, pre-swizzled into WMMA B-fragment layout.
// B 32x16 f16 fragment: lane l covers col N=l%16; lanes 0-15 hold K=0..7,16..23,
// lanes 16-31 hold K=8..15,24..31; dword d packs (K, K+1) in that order.
// W0 keeps only the 9 state-feature rows (temb rows folded into the ts table).
// ============================================================================
__global__ void prep_pack(const float* Wl0, const float* Wl1, const float* Wl2,
                          const float* bl0, const float* bl1, const float* bl2,
                          const float* Wpos, const float* Wvel, const float* Wen,
                          const float* Wunc, const float* bpos, const float* bvel,
                          const float* ben, const float* bunc, unsigned int* ws) {
  int i = blockIdx.x * blockDim.x + threadIdx.x;
  if (i >= OFF_PE) return;
  if (i < OFF_WH) {
    const float* W; int din, KT, base;
    if (i < OFF_W1)      { W = Wl0; din = 9;   KT = 1; base = OFF_W0; }
    else if (i < OFF_W2) { W = Wl1; din = 128; KT = 4; base = OFF_W1; }
    else                 { W = Wl2; din = 128; KT = 4; base = OFF_W2; }
    int f = i - base;
    int d = f & 7, lane = (f >> 3) & 31, q = f >> 8;
    int kt = q % KT; q /= KT;
    int gate = q % 3, wave = q / 3;
    int ntile = (gate == 0) ? wave : (gate == 1) ? 16 + wave : 24 + wave;
    int col = ntile * 16 + (lane & 15);
    int k = kt * 32 + ((lane & 16) ? 8 : 0) + ((d & 4) ? 16 : 0) + (d & 3) * 2;
    float lo = (k     < din) ? W[k * 512 + col]       : 0.f;
    float hi = (k + 1 < din) ? W[(k + 1) * 512 + col] : 0.f;
    ws[i] = pack2h(lo, hi);
  } else if (i < OFF_BL) {                    // fused head matrix [128 x 16]
    int f = i - OFF_WH;
    int d = f & 7, lane = (f >> 3) & 31, kt = f >> 8;
    int col = lane & 15;
    int k = kt * 32 + ((lane & 16) ? 8 : 0) + ((d & 4) ? 16 : 0) + (d & 3) * 2;
    float v[2];
    #pragma unroll
    for (int s = 0; s < 2; ++s) {
      int kk = k + s; float x;
      if (col < 2)       x = Wpos[kk * 2 + col];
      else if (col < 4)  x = Wvel[kk * 2 + (col - 2)];
      else if (col < 6)  x = Wen [kk * 2 + (col - 4)];
      else if (col < 14) x = Wunc[kk * 8 + (col - 6)];
      else               x = 0.f;
      v[s] = x;
    }
    ws[i] = pack2h(v[0], v[1]);
  } else if (i < OFF_BH) {                    // LSTM biases (f32 copy)
    int j = i - OFF_BL; int L = j >> 9; int c = j & 511;
    const float* b = (L == 0) ? bl0 : (L == 1) ? bl1 : bl2;
    ((float*)ws)[i] = b[c];
  } else {                                    // fused head bias [16]
    int j = i - OFF_BH;
    float x = (j < 2) ? bpos[j] : (j < 4) ? bvel[j - 2] : (j < 6) ? ben[j - 4]
            : (j < 14) ? bunc[j - 6] : 0.f;
    ((float*)ws)[i] = x;
  }
}

// ============================================================================
// P1b: per-step temb contribution to layer-0 gates (same for all 1024 rows):
//      ts[t][col] = sum_j temb[t][j] * Wl0[9+j][col]
// ============================================================================
__global__ void prep_ts(const float* temb, const float* Wl0, float* wsf) {
  int i = blockIdx.x * blockDim.x + threadIdx.x;
  if (i >= T_STEPS * 512) return;
  int t = i >> 9, col = i & 511;
  float s = 0.f;
  for (int j = 0; j < 32; ++j) s += temb[t * 32 + j] * Wl0[(9 + j) * 512 + col];
  wsf[OFF_TS + i] = s;
}

// ============================================================================
// P2: physics encoder  rules[B,4] -> relu -> relu -> pe[B,64]
// ============================================================================
__global__ void prep_pe(const float* grav, const float* fric, const float* elas,
                        const float* damp, const float* W1, const float* b1,
                        const float* W2, const float* b2, float* wsf) {
  __shared__ float h[256];
  int b = blockIdx.x, j = threadIdx.x;
  float r0 = grav[b], r1 = fric[b], r2 = elas[b], r3 = damp[b];
  #pragma unroll
  for (int q = 0; q < 4; ++q) {
    int c = j * 4 + q;
    float v = b1[c] + r0 * W1[c] + r1 * W1[256 + c] + r2 * W1[512 + c] + r3 * W1[768 + c];
    h[c] = v > 0.f ? v : 0.f;
  }
  __syncthreads();
  float acc = b2[j];
  for (int k = 0; k < 256; ++k) acc += h[k] * W2[k * 64 + j];
  wsf[OFF_PE + b * 64 + j] = acc > 0.f ? acc : 0.f;
}

// ============================================================================
// P3: loop-invariant per-layer terms: g = sigmoid(pe@Wg+bg), tr = tanh(pe@Wt+bt),
//     att = (pe@Wv+bv) @ Wo + bo      (seq-len-1 attention). One block per row.
// ============================================================================
__global__ void prep_gat(const float* Wv, const float* bv, const float* Wo,
                         const float* bo, const float* Wg, const float* bg,
                         const float* Wt, const float* bt, int layer, void* wsp) {
  const float* wsf = (const float*)wsp;
  _Float16* gat = (_Float16*)((unsigned int*)wsp + OFF_GAT);
  __shared__ float pe[64];
  __shared__ float v[128];
  int n = blockIdx.x, j = threadIdx.x;
  if (j < 64) pe[j] = wsf[OFF_PE + (n >> 2) * 64 + j];
  __syncthreads();
  float av = bv[j], gg = bg[j], tt = bt[j];
  for (int p = 0; p < 64; ++p) {
    float pv = pe[p];
    av += pv * Wv[p * 128 + j];
    gg += pv * Wg[p * 128 + j];
    tt += pv * Wt[p * 128 + j];
  }
  v[j] = av;
  __syncthreads();
  float aa = bo[j];
  for (int m = 0; m < 128; ++m) aa += v[m] * Wo[m * 128 + j];
  int base = (layer * 3) * 1024 * 128 + n * 128 + j;
  gat[base]                  = (_Float16)fsigmoid(gg);   // g
  gat[base + 1024 * 128]     = (_Float16)aa;             // att
  gat[base + 2 * 1024 * 128] = (_Float16)ftanh(tt);      // trans
}

// ============================================================================
// Main kernel: one workgroup per 16-row tile, 300 sequential steps, WMMA f16.
// Weights: 220 KB dynamic LDS. Loop-invariant scalars (gate biases, physics
// g/a/tr terms) live in registers; layer-0 A-fragment is built in registers
// straight from the padded state tile (no staging, no extra barrier).
// ============================================================================
static __device__ __forceinline__ v16h load_a(const _Float16* rowp, int kbase, int lane) {
  int off = kbase + ((lane & 16) ? 8 : 0);
  union { uint4 u[2]; v16h v; } t;
  t.u[0] = *(const uint4*)(rowp + off);
  t.u[1] = *(const uint4*)(rowp + off + 16);
  return t.v;
}

static __device__ __forceinline__ v16h load_b(const unsigned int* p) {
  union { uint4 u[2]; v16h v; } t;
  t.u[0] = ((const uint4*)p)[0];
  t.u[1] = ((const uint4*)p)[1];
  return t.v;
}

static __device__ __forceinline__ v8f splat8(float b) {
  v8f c;
  #pragma unroll
  for (int e = 0; e < 8; ++e) c[e] = b;
  return c;
}

template <int KT>
static __device__ __forceinline__ v8f gate_mm(const v16h* afr, const unsigned int* wbg,
                                              float bias, int lane) {
  v16h bf[KT];
  #pragma unroll
  for (int kt = 0; kt < KT; ++kt) bf[kt] = load_b(wbg + kt * 256 + lane * 8);
  v8f c = splat8(bias);
  #pragma unroll
  for (int kt = 0; kt < KT; ++kt)
    c = __builtin_amdgcn_wmma_f32_16x16x32_f16(false, afr[kt], false, bf[kt],
                                               (short)0, c, false, false);
  return c;
}

template <int KT>
static __device__ __forceinline__ void lstm_layer(const v16h* afr, const unsigned int* wb,
                                                  const float (&bias3)[3],
                                                  const float (&g8)[8], const float (&a8)[8],
                                                  const float (&t8)[8],
                                                  _Float16* xout, int wave, int lane) {
  const int nlo = lane & 15;
  const int hi8 = (lane & 16) ? 8 : 0;
  v8f ai = gate_mm<KT>(afr, wb + (wave * 3 + 0) * KT * 256, bias3[0], lane);
  v8f ac = gate_mm<KT>(afr, wb + (wave * 3 + 1) * KT * 256, bias3[1], lane);
  v8f ao = gate_mm<KT>(afr, wb + (wave * 3 + 2) * KT * 256, bias3[2], lane);
  #pragma unroll
  for (int e = 0; e < 8; ++e) {
    int row = e + hi8;
    int col = wave * 16 + nlo;
    float cn = fsigmoid(ai[e]) * ftanh(ac[e]);
    float hc = fsigmoid(ao[e]) * ftanh(cn);
    float gg = g8[e];
    xout[row * 128 + col] = (_Float16)(hc * gg + a8[e] * (1.f - gg) + t8[e]);
  }
}

__global__ __launch_bounds__(256) void traj_main(const float* ic,
                                                 const unsigned int* ws, float* out) {
  extern __shared__ unsigned int s_w[];    // 220 KB: W0|W1|W2|WH fragment dwords
  __shared__ _Float16 s_x[2][16][128];     // double-buffered activations
  __shared__ float    s_state[16][16];     // padded: [9..15] stay zero
  __shared__ float    s_head[16][16];

  const int tid  = threadIdx.x;
  const int lane = tid & 31, wave = tid >> 5;
  const int nlo  = lane & 15;
  const int hi8  = (lane & 16) ? 8 : 0;
  const int r0   = blockIdx.x * 16;
  const float* wsf = (const float*)ws;
  const _Float16* gatg = (const _Float16*)(ws + OFF_GAT);

  // ---- one-time LDS fill: all weight fragments (amortized over 300 steps)
  for (int i = tid; i < FRAG_DWORDS / 4; i += 256)
    ((uint4*)s_w)[i] = ((const uint4*)ws)[i];
  for (int i = tid; i < 256; i += 256) {
    int row = i >> 4, k = i & 15;
    s_state[row][k] = (k < 9) ? ic[(r0 + row) * 9 + k] : 0.f;
  }

  // ---- loop-invariant per-lane registers: g/a/tr terms and biases
  float gv[9][8];                       // [layer*3 + {g,a,tr}][element]
  #pragma unroll
  for (int a = 0; a < 9; ++a)
    #pragma unroll
    for (int e = 0; e < 8; ++e)
      gv[a][e] = (float)gatg[(a * 1024 + r0 + e + hi8) * 128 + wave * 16 + nlo];

  float bl3[3][3];                      // [layer][gate i/c/o]
  #pragma unroll
  for (int L = 0; L < 3; ++L)
    #pragma unroll
    for (int gi = 0; gi < 3; ++gi) {
      int ntile = (gi == 0) ? wave : (gi == 1) ? 16 + wave : 24 + wave;
      bl3[L][gi] = wsf[OFF_BL + L * 512 + ntile * 16 + nlo];
    }
  float bh = wsf[OFF_BH + nlo];
  __syncthreads();

  for (int t = 0; t < T_STEPS; ++t) {
    float t_frac = (float)t * (1.0f / (float)T_STEPS);
    // per-step temb contribution for this lane's three gate columns (L2-resident)
    int tsrow = OFF_TS + t * 512 + wave * 16 + nlo;
    float eb[3] = { bl3[0][0] + wsf[tsrow],
                    bl3[0][1] + wsf[tsrow + 256],
                    bl3[0][2] + wsf[tsrow + 384] };
    int tsn = OFF_TS + ((t + 1 < T_STEPS) ? t + 1 : t) * 512 + wave * 16 + nlo;
    __builtin_prefetch(&wsf[tsn], 0, 0);          // global_prefetch_b8

    { // layer 0: A-fragment built in registers from padded state (K=32)
      int soff = (lane & 16) ? 8 : 0;
      float4 f0 = *(const float4*)&s_state[nlo][soff];
      float4 f1 = *(const float4*)&s_state[nlo][soff + 4];
      union { uint4 u[2]; v16h v; } a0;
      a0.u[0].x = pack2h(f0.x, f0.y);
      a0.u[0].y = pack2h(f0.z, f0.w);
      a0.u[0].z = pack2h(f1.x, f1.y);
      a0.u[0].w = pack2h(f1.z, f1.w);
      a0.u[1].x = 0u; a0.u[1].y = 0u; a0.u[1].z = 0u; a0.u[1].w = 0u;
      v16h af0[1] = { a0.v };
      lstm_layer<1>(af0, s_w + OFF_W0, eb, gv[0], gv[1], gv[2],
                    &s_x[0][0][0], wave, lane);
    }
    __syncthreads();
    { // layer 1
      const _Float16* rp = &s_x[0][nlo][0];
      v16h af[4] = { load_a(rp, 0, lane), load_a(rp, 32, lane),
                     load_a(rp, 64, lane), load_a(rp, 96, lane) };
      lstm_layer<4>(af, s_w + OFF_W1, bl3[1], gv[3], gv[4], gv[5],
                    &s_x[1][0][0], wave, lane);
    }
    __syncthreads();
    { // layer 2
      const _Float16* rp = &s_x[1][nlo][0];
      v16h af[4] = { load_a(rp, 0, lane), load_a(rp, 32, lane),
                     load_a(rp, 64, lane), load_a(rp, 96, lane) };
      lstm_layer<4>(af, s_w + OFF_W2, bl3[2], gv[6], gv[7], gv[8],
                    &s_x[0][0][0], wave, lane);
    }
    __syncthreads();

    if (wave == 0) { // fused heads + state update, all inside wave 0
      const _Float16* rp = &s_x[0][nlo][0];
      v16h bfh[4];
      #pragma unroll
      for (int kt = 0; kt < 4; ++kt)
        bfh[kt] = load_b(s_w + OFF_WH + (kt * 32 + lane) * 8);
      v8f c = splat8(bh);
      #pragma unroll
      for (int kt = 0; kt < 4; ++kt) {
        v16h af = load_a(rp, kt * 32, lane);
        c = __builtin_amdgcn_wmma_f32_16x16x32_f16(false, af, false, bfh[kt],
                                                   (short)0, c, false, false);
      }
      #pragma unroll
      for (int e = 0; e < 8; ++e) s_head[e + hi8][nlo] = c[e];
      // same-wave DS ops are in-order: lanes 0-15 can read the transpose back
      if (lane < 16) {
        int row = lane, n = r0 + row, bb = n >> 2, ball = n & 3;
        float* st = s_state[row];
        st[0]  = t_frac;
        st[1] += s_head[row][0]; st[2] += s_head[row][1];
        st[3] += s_head[row][2]; st[4] += s_head[row][3];
        st[7]  = s_head[row][4]; st[8]  = s_head[row][5];
        int tb = (bb * T_STEPS + t) * 4 + ball;
        float* tp = out + tb * 9;
        #pragma unroll
        for (int k = 0; k < 9; ++k) tp[k] = st[k];
        float* up = out + OUT_UNC_BASE + tb * 8;
        #pragma unroll
        for (int k = 0; k < 8; ++k) up[k] = fsoftplus(s_head[row][6 + k]);
      }
    }
    __syncthreads();
  }
}

// ============================================================================
extern "C" void kernel_launch(void* const* d_in, const int* in_sizes, int n_in,
                              void* d_out, int out_size, void* d_ws, size_t ws_size,
                              hipStream_t stream) {
  (void)in_sizes; (void)n_in; (void)out_size; (void)ws_size;
  const float* IC   = (const float*)d_in[0];
  const float* GRAV = (const float*)d_in[1];
  const float* FRIC = (const float*)d_in[2];
  const float* ELAS = (const float*)d_in[3];
  const float* DAMP = (const float*)d_in[4];
  const float* PEW1 = (const float*)d_in[5];
  const float* PEB1 = (const float*)d_in[6];
  const float* PEW2 = (const float*)d_in[7];
  const float* PEB2 = (const float*)d_in[8];
  const float* TEMB = (const float*)d_in[9];
  // layers: base 10 + 10*L : Wl, bl, Wv, bv, Wo, bo, Wg, bg, Wt, bt
  const float* WL[3]; const float* BL[3];
  const float* WV[3]; const float* BV[3]; const float* WO[3]; const float* BO[3];
  const float* WG[3]; const float* BG[3]; const float* WT[3]; const float* BT[3];
  for (int L = 0; L < 3; ++L) {
    int b = 10 + 10 * L;
    WL[L] = (const float*)d_in[b + 0]; BL[L] = (const float*)d_in[b + 1];
    WV[L] = (const float*)d_in[b + 2]; BV[L] = (const float*)d_in[b + 3];
    WO[L] = (const float*)d_in[b + 4]; BO[L] = (const float*)d_in[b + 5];
    WG[L] = (const float*)d_in[b + 6]; BG[L] = (const float*)d_in[b + 7];
    WT[L] = (const float*)d_in[b + 8]; BT[L] = (const float*)d_in[b + 9];
  }
  const float* WPOS = (const float*)d_in[40]; const float* BPOS = (const float*)d_in[41];
  const float* WVEL = (const float*)d_in[42]; const float* BVEL = (const float*)d_in[43];
  const float* WEN  = (const float*)d_in[44]; const float* BEN  = (const float*)d_in[45];
  const float* WUNC = (const float*)d_in[46]; const float* BUNC = (const float*)d_in[47];

  unsigned int* ws = (unsigned int*)d_ws;

  prep_pack<<<(OFF_PE + 255) / 256, 256, 0, stream>>>(
      WL[0], WL[1], WL[2], BL[0], BL[1], BL[2],
      WPOS, WVEL, WEN, WUNC, BPOS, BVEL, BEN, BUNC, ws);

  prep_ts<<<(T_STEPS * 512 + 255) / 256, 256, 0, stream>>>(TEMB, WL[0], (float*)d_ws);

  prep_pe<<<256, 64, 0, stream>>>(GRAV, FRIC, ELAS, DAMP, PEW1, PEB1, PEW2, PEB2,
                                  (float*)d_ws);

  for (int L = 0; L < 3; ++L)
    prep_gat<<<1024, 128, 0, stream>>>(WV[L], BV[L], WO[L], BO[L], WG[L], BG[L],
                                       WT[L], BT[L], L, d_ws);

  traj_main<<<64, 256, FRAG_DWORDS * 4, stream>>>(IC, ws, (float*)d_out);
}
